// CRF_31181462569486
// MI455X (gfx1250) — matrix operand order repaired and checked by reading, not compiled, not run on hardware
//
#include <hip/hip_runtime.h>
#include <hip/hip_bf16.h>

typedef __attribute__((ext_vector_type(16))) __bf16 v16bf;
typedef __attribute__((ext_vector_type(8)))  float  v8f;

#define Bn 512
#define Mn 512
#define Dn 128
#define Ln 26
#define LP 32   // padded label dim

__device__ __forceinline__ float readlane_f(float v, int l) {
  return __builtin_bit_cast(float, __builtin_amdgcn_readlane(__builtin_bit_cast(int, v), l));
}

// Pack 8 f32 into bf16 hi + bf16 residual-lo halves of a v16bf operand.
__device__ __forceinline__ void pack8(const float4& u, const float4& v,
                                      v16bf& h, v16bf& l, int base) {
  float t[8] = {u.x, u.y, u.z, u.w, v.x, v.y, v.z, v.w};
#pragma unroll
  for (int i = 0; i < 8; ++i) {
    __bf16 hh = (__bf16)t[i];
    h[base + i] = hh;
    l[base + i] = (__bf16)(t[i] - (float)hh);
  }
}

// ---------------------------------------------------------------------------
// Kernel 1: e[b,m,l] = <x[b,m,:], W[l,:]> via bf16x3 WMMA (near-f32 accuracy).
// Each wave: 4 row-tiles (64 tokens) x 32 labels; B fragments amortized 4x.
// eP columns 26..31 hold junk (clamped W rows) -- consumers mask lanes >= 26.
// ---------------------------------------------------------------------------
__global__ void __launch_bounds__(256)
crf_emis_wmma(const float* __restrict__ x, const float* __restrict__ W,
              float* __restrict__ eP) {
  const int  lane = threadIdx.x & 31;
  const int  wid  = threadIdx.x >> 5;
  const long g    = (long)blockIdx.x * 8 + wid;   // wave index: 64 tokens each
  const int  hi16 = lane >> 4;
  const int  l15  = lane & 15;
  const long row0 = g * 64 + l15;                 // A row of tile r = row0 + r*16

  v8f acc[4][2] = {};

#pragma unroll
  for (int kt = 0; kt < 4; ++kt) {
    const int kbase = kt * 32;
    const int koff  = kbase + hi16 * 16;          // B: lane half selects K 0-15 / 16-31

    // ---- B tile cols 0..15 (always valid rows of W)
    v16bf b0h, b0l;
    {
      const float* wr = W + (long)l15 * Dn + koff;
      pack8(*(const float4*)(wr),     *(const float4*)(wr + 4),  b0h, b0l, 0);
      pack8(*(const float4*)(wr + 8), *(const float4*)(wr + 12), b0h, b0l, 8);
    }
    // ---- B tile cols 16..31: clamp rows >= 26 (junk output cols are masked later)
    v16bf b1h, b1l;
    {
      const int col  = 16 + l15;
      const int wrow = (col < Ln) ? col : 0;
      const float* wr = W + (long)wrow * Dn + koff;
      pack8(*(const float4*)(wr),     *(const float4*)(wr + 4),  b1h, b1l, 0);
      pack8(*(const float4*)(wr + 8), *(const float4*)(wr + 12), b1h, b1l, 8);
    }

#pragma unroll
    for (int r = 0; r < 4; ++r) {
      // ---- A (16x32 bf16): lanes 0-15 hold K 0-15, lanes 16-31 K 16-31
      const float* xr = x + (row0 + r * 16) * Dn;
      const int ka = kbase + hi16 * 16;
      const int kb = ka + 8;
      v16bf ah, al;
      pack8(*(const float4*)(xr + ka), *(const float4*)(xr + ka + 4), ah, al, 0);
      pack8(*(const float4*)(xr + kb), *(const float4*)(xr + kb + 4), ah, al, 8);

      acc[r][0] = __builtin_amdgcn_wmma_f32_16x16x32_bf16(false, ah, false, b0h, (short)0, acc[r][0], false, false);
      acc[r][0] = __builtin_amdgcn_wmma_f32_16x16x32_bf16(false, al, false, b0h, (short)0, acc[r][0], false, false);
      acc[r][0] = __builtin_amdgcn_wmma_f32_16x16x32_bf16(false, ah, false, b0l, (short)0, acc[r][0], false, false);
      acc[r][1] = __builtin_amdgcn_wmma_f32_16x16x32_bf16(false, ah, false, b1h, (short)0, acc[r][1], false, false);
      acc[r][1] = __builtin_amdgcn_wmma_f32_16x16x32_bf16(false, al, false, b1h, (short)0, acc[r][1], false, false);
      acc[r][1] = __builtin_amdgcn_wmma_f32_16x16x32_bf16(false, ah, false, b1l, (short)0, acc[r][1], false, false);
    }
  }

  // C layout: VGPR rr -> (M = rr + hi16*8, N = l15)
#pragma unroll
  for (int r = 0; r < 4; ++r) {
    float* er = eP + (g * 64 + r * 16 + hi16 * 8) * LP;
#pragma unroll
    for (int rr = 0; rr < 8; ++rr) {
      er[(long)rr * LP + l15]      = acc[r][0][rr];
      er[(long)rr * LP + 16 + l15] = acc[r][1][rr];
    }
  }
}

// ---------------------------------------------------------------------------
// Kernel 2: per-batch forward scan + gold-path score. One wave per batch.
// alpha'_j = log( sum_k exp(alphaRel_k + e_k) * expT[k][j] ), recentred by
// lane 0's value each step; e rows streamed through a 16-deep register
// window so loads issue ~500 cycles ahead of the serial dependency chain.
// ---------------------------------------------------------------------------
__global__ void __launch_bounds__(32)
crf_scan(const float* __restrict__ eP, const int* __restrict__ y,
         const float* __restrict__ T, float* __restrict__ logp) {
  const int  b    = blockIdx.x;
  const int  lane = threadIdx.x;
  const bool act  = lane < Ln;
  const int  lcl  = act ? lane : 0;    // clamped index: keep inactive-lane loads in bounds

  // Precompute exp(T[k][lane]) into registers (column of expT per lane).
  float col[Ln];
#pragma unroll
  for (int k = 0; k < Ln; ++k)
    col[k] = act ? __expf(T[k * Ln + lcl]) : 0.0f;

  const float* eB = eP + (long)b * Mn * LP;
  float alphaRel = 0.0f;   // lanes >= 26 collapse to -inf after step 1 (col[]=0)
  float shift    = 0.0f;   // wave-uniform accumulated offset

  auto step = [&](float e_cur) {
    const float v = alphaRel + e_cur;
    const float p = __expf(v);
    float s0 = 0.0f, s1 = 0.0f, s2 = 0.0f, s3 = 0.0f;   // break the FMA chain 4-way
#pragma unroll
    for (int k = 0; k < Ln; k += 4) {
      s0 = fmaf(readlane_f(p, k),     col[k],     s0);
      s1 = fmaf(readlane_f(p, k + 1), col[k + 1], s1);
      if (k + 2 < Ln) s2 = fmaf(readlane_f(p, k + 2), col[k + 2], s2);
      if (k + 3 < Ln) s3 = fmaf(readlane_f(p, k + 3), col[k + 3], s3);
    }
    const float a  = __logf((s0 + s1) + (s2 + s3));
    const float a0 = readlane_f(a, 0);
    alphaRel = a - a0;
    shift += a0;
  };

  // 16-deep load window: ev[i] holds e[:, base + i].
  float ev[16];
#pragma unroll
  for (int i = 0; i < 16; ++i)
    ev[i] = eB[i * LP + lane];

  // Main: 31 x 16 = 496 steps; prefetch row t+16 (max row index 511, in bounds).
  for (int outer = 0; outer < 31; ++outer) {
    const int base = outer * 16;
#pragma unroll
    for (int i = 0; i < 16; ++i) {
      const float cur = ev[i];
      ev[i] = eB[(base + 16 + i) * LP + lane];
      step(cur);
    }
  }
  // Tail: steps 496..510 consume ev[0..14]; ev[15] = e[:, 511] for logZ.
#pragma unroll
  for (int i = 0; i < 15; ++i)
    step(ev[i]);

  // logZ = shift + LSE(alphaRel + e[:, M-1])
  {
    const float v = act ? alphaRel + ev[15] : -INFINITY;
    float m = v;
#pragma unroll
    for (int off = 16; off > 0; off >>= 1)
      m = fmaxf(m, __shfl_xor(m, off));
    float s = act ? __expf(v - m) : 0.0f;
#pragma unroll
    for (int off = 16; off > 0; off >>= 1)
      s += __shfl_xor(s, off);
    const float logZ = shift + m + __logf(s);

    // Gold path: sum_t e[t, y_t] + sum_t T[y_t, y_{t+1}]
    const int* yb = y + (long)b * Mn;
    float acc = 0.0f;
    for (int t = lane; t < Mn; t += 32)
      acc += eB[t * LP + yb[t]];
    for (int t = lane; t < Mn - 1; t += 32)
      acc += T[yb[t] * Ln + yb[t + 1]];
#pragma unroll
    for (int off = 16; off > 0; off >>= 1)
      acc += __shfl_xor(acc, off);

    if (lane == 0) logp[b] = acc - logZ;
  }
}

// ---------------------------------------------------------------------------
// Kernel 3: nll = -sum_b logp[b] / B   (deterministic tree reduction)
// ---------------------------------------------------------------------------
__global__ void __launch_bounds__(512)
crf_reduce(const float* __restrict__ logp, float* __restrict__ out) {
  __shared__ float sm[512];
  const int t = threadIdx.x;
  sm[t] = logp[t];
  __syncthreads();
#pragma unroll
  for (int s = 256; s > 0; s >>= 1) {
    if (t < s) sm[t] += sm[t + s];
    __syncthreads();
  }
  if (t == 0) out[0] = -sm[0] / (float)Bn;
}

extern "C" void kernel_launch(void* const* d_in, const int* in_sizes, int n_in,
                              void* d_out, int out_size, void* d_ws, size_t ws_size,
                              hipStream_t stream) {
  const float* x = (const float*)d_in[0];   // [B, M, D]
  const int*   y = (const int*)d_in[1];     // [B, M]
  const float* W = (const float*)d_in[2];   // [L, D]
  const float* T = (const float*)d_in[3];   // [L, L]

  float* eP   = (float*)d_ws;                  // [B*M, 32] padded emissions
  float* logp = eP + (long)Bn * Mn * LP;       // [B]

  // 16384 row-tiles of 16 tokens; 4 tiles per wave; 8 waves per block.
  crf_emis_wmma<<<(Bn * Mn / 64) / 8, 256, 0, stream>>>(x, W, eP);
  crf_scan<<<Bn, 32, 0, stream>>>(eP, y, T, logp);
  crf_reduce<<<1, 512, 0, stream>>>(logp, (float*)d_out);
}
